// MixtralLayer_85341000171522
// MI455X (gfx1250) — matrix operand order repaired and checked by reading, compiled, not run on hardware
//
#include <hip/hip_runtime.h>

#define DEVI __device__ __forceinline__

typedef __bf16 v16bf __attribute__((ext_vector_type(16)));
typedef float  v8f   __attribute__((ext_vector_type(8)));
typedef unsigned int   u32;
typedef unsigned short u16;
typedef u32 u32x4 __attribute__((ext_vector_type(4)));
typedef u32 u32x2 __attribute__((ext_vector_type(2)));

union Frag { v16bf v; u32x4 q[2]; u16 h[16]; };

constexpr int T = 512, D = 2048, H = 32, KVH = 8, DH = 64;
constexpr int NQ = (H + 2 * KVH) * DH;      // 3072
constexpr int FF = 7168, NE = 8;
constexpr float EPS = 1e-5f;
constexpr int RC = 1280;                    // MoE row capacity (<= 1024 + 8*15 padded)

// ---- workspace layout (bytes) ----
constexpr size_t OFF_RES1 = 0;
constexpr size_t OFF_X1  = OFF_RES1 + (size_t)T * D * 4;
constexpr size_t OFF_QKV = OFF_X1  + (size_t)T * D * 2;
constexpr size_t OFF_QR  = OFF_QKV + (size_t)T * NQ * 4;
constexpr size_t OFF_KR  = OFF_QR  + (size_t)H * T * DH * 2;
constexpr size_t OFF_VT  = OFF_KR  + (size_t)KVH * T * DH * 2;
constexpr size_t OFF_ATT = OFF_VT  + (size_t)KVH * T * DH * 2;
constexpr size_t OFF_X2  = OFF_ATT + (size_t)T * D * 2;
constexpr size_t OFF_TW  = OFF_X2  + (size_t)T * D * 2;
constexpr size_t OFF_TI  = OFF_TW  + (size_t)T * 2 * 4;
constexpr size_t OFF_RO  = OFF_TI  + (size_t)T * 2 * 4;
constexpr size_t OFF_LS  = OFF_RO  + (size_t)T * 2 * 4;
constexpr size_t OFF_CP  = OFF_LS  + (size_t)NE * T * 4;
constexpr size_t OFF_BS  = OFF_CP  + 64;
constexpr size_t OFF_ACT = ((OFF_BS + 64 + 255) / 256) * 256;
constexpr size_t OFF_EO  = OFF_ACT + (size_t)RC * FF * 2;

DEVI u16 f2bf(float f) {
  u32 u = __builtin_bit_cast(u32, f);
  u32 r = (u + 0x7FFFu + ((u >> 16) & 1u)) >> 16;   // RNE
  return (u16)r;
}
DEVI float bf2f(u16 s) { return __builtin_bit_cast(float, (u32)s << 16); }
DEVI v8f v8zero() { v8f z; for (int i = 0; i < 8; i++) z[i] = 0.f; return z; }

DEVI v8f wmma_bf(const Frag& a, const Frag& b, v8f c) {
  return __builtin_amdgcn_wmma_f32_16x16x32_bf16(false, a.v, false, b.v,
                                                 (short)0, c, false, false);
}

// ---- async global->LDS copy (CDNA5 GLOBAL_LOAD_ASYNC_TO_LDS_B32), guarded ----
#if __has_builtin(__builtin_amdgcn_global_load_async_to_lds_b32) && \
    __has_builtin(__builtin_amdgcn_s_wait_asynccnt)
#define USE_ASYNC_LDS 1
#else
#define USE_ASYNC_LDS 0
#endif

typedef __attribute__((address_space(1))) int gint;
typedef __attribute__((address_space(3))) int lint;

DEVI void async_cp_b32(const u16* gsrc, u16* ldst) {
#if USE_ASYNC_LDS
  __builtin_amdgcn_global_load_async_to_lds_b32((gint*)gsrc, (lint*)ldst, 0, 0);
#else
  *(u32*)ldst = *(const u32*)gsrc;
#endif
}
DEVI void async_wait() {
#if USE_ASYNC_LDS
  __builtin_amdgcn_s_wait_asynccnt(0);
#endif
}

// ---- XOR reductions across the 16-lane half via immediate ds_swizzle ---------
template <int IMM>
DEVI float swzf(float x) {
  return __builtin_bit_cast(float,
      __builtin_amdgcn_ds_swizzle(__builtin_bit_cast(int, x), IMM));
}
DEVI float redmax16(float x) {
  x = fmaxf(x, swzf<0x041F>(x));   // xor 1
  x = fmaxf(x, swzf<0x081F>(x));   // xor 2
  x = fmaxf(x, swzf<0x101F>(x));   // xor 4
  x = fmaxf(x, swzf<0x201F>(x));   // xor 8
  return x;
}
DEVI float redsum16(float x) {
  x += swzf<0x041F>(x);
  x += swzf<0x081F>(x);
  x += swzf<0x101F>(x);
  x += swzf<0x201F>(x);
  return x;
}

// ---------------- residual add + RMSNorm (also used as plain RMSNorm) ----------
__global__ __launch_bounds__(256) void k_addnorm(const float* __restrict__ x,
                                                 const float* __restrict__ add,
                                                 const float* __restrict__ w,
                                                 float* __restrict__ res_out,
                                                 u16* __restrict__ nrm_out) {
  int t = blockIdx.x, tid = threadIdx.x;
  __shared__ float red[256];
  float v[8]; float ss = 0.f;
#pragma unroll
  for (int i = 0; i < 8; i++) {
    int d = tid + i * 256;
    float a = x[(size_t)t * D + d];
    if (add) a += add[(size_t)t * D + d];
    v[i] = a; ss += a * a;
    if (res_out) res_out[(size_t)t * D + d] = a;
  }
  red[tid] = ss; __syncthreads();
  for (int s = 128; s > 0; s >>= 1) { if (tid < s) red[tid] += red[tid + s]; __syncthreads(); }
  float rs = rsqrtf(red[0] / (float)D + EPS);
#pragma unroll
  for (int i = 0; i < 8; i++) {
    int d = tid + i * 256;
    nrm_out[(size_t)t * D + d] = f2bf(v[i] * rs * w[d]);
  }
}

// ---------------- dense GEMM: C[M,N] = Abf16[M,K] * W[N,K]^T ------------------
// grid (M/16, N/256), 256 threads = 8 waves, wave w owns cols [w*32, w*32+32)
template <int RESADD>
__global__ __launch_bounds__(256) void k_gemm(const u16* __restrict__ A,
                                              const float* __restrict__ Bw,
                                              float* __restrict__ Cout,
                                              const float* __restrict__ resid,
                                              int Kd, int N) {
  int m0 = blockIdx.x * 16, n0 = blockIdx.y * 256;
  int tid = threadIdx.x, lane = tid & 31, w = tid >> 5;
  __shared__ __align__(16) u16 AT[16 * 32];
  __shared__ __align__(16) u16 BT[256 * 32];
  v8f acc[2] = { v8zero(), v8zero() };
  for (int k0 = 0; k0 < Kd; k0 += 32) {
    { int e = tid * 2, r = e >> 5, kk = e & 31;
      async_cp_b32(&A[(size_t)(m0 + r) * Kd + k0 + kk], &AT[r * 32 + kk]); }
    { const float* src = Bw + (size_t)(n0 + tid) * Kd + k0;
      if (k0 + 32 < Kd) __builtin_prefetch(src + 32, 0, 1);
#pragma unroll
      for (int i = 0; i < 8; i++) {
        float4 f = *(const float4*)(src + i * 4);
        u32x2 p; p.x = (u32)f2bf(f.x) | ((u32)f2bf(f.y) << 16);
        p.y = (u32)f2bf(f.z) | ((u32)f2bf(f.w) << 16);
        *(u32x2*)&BT[tid * 32 + i * 4] = p;
      } }
    async_wait();
    __syncthreads();
    int row = lane & 15, k8 = (lane >> 4) << 3, kb16 = (lane >> 4) << 4;
    Frag fa; fa.q[0] = *(const u32x4*)&AT[row * 32 + kb16];
    fa.q[1] = *(const u32x4*)&AT[row * 32 + kb16 + 8];
#pragma unroll
    for (int c = 0; c < 2; c++) {
      int n = w * 32 + c * 16 + (lane & 15);
      Frag fb; fb.q[0] = *(const u32x4*)&BT[n * 32 + kb16];
      fb.q[1] = *(const u32x4*)&BT[n * 32 + kb16 + 8];
      acc[c] = wmma_bf(fa, fb, acc[c]);
    }
    __syncthreads();
  }
  int rb = (lane >> 4) << 3;
#pragma unroll
  for (int c = 0; c < 2; c++)
#pragma unroll
    for (int r = 0; r < 8; r++) {
      int gm = m0 + rb + r, gn = n0 + w * 32 + c * 16 + (lane & 15);
      float val = acc[c][r];
      if (RESADD) val += resid[(size_t)gm * N + gn];
      Cout[(size_t)gm * N + gn] = val;
    }
}

// ---------------- RoPE + layout shuffle (Q:(H,T,DH) K:(KV,T,DH) V:(KV,DH,T)) --
__global__ __launch_bounds__(64) void k_rope(const float* __restrict__ qkv,
                                             const float* __restrict__ cs,
                                             const float* __restrict__ sn,
                                             u16* __restrict__ qr,
                                             u16* __restrict__ kr,
                                             u16* __restrict__ vt) {
  int t = blockIdx.x, u = blockIdx.y, d = threadIdx.x;
  if (u < H) {
    const float* s = qkv + (size_t)t * NQ + u * DH;
    float o;
    if (d < 32) { float c = cs[t * 32 + d], si = sn[t * 32 + d]; o = s[d] * c - s[d + 32] * si; }
    else { int dd = d - 32; float c = cs[t * 32 + dd], si = sn[t * 32 + dd]; o = s[d] * c + s[dd] * si; }
    qr[((size_t)u * T + t) * DH + d] = f2bf(o);
  } else if (u < H + KVH) {
    int kv = u - H;
    const float* s = qkv + (size_t)t * NQ + H * DH + kv * DH;
    float o;
    if (d < 32) { float c = cs[t * 32 + d], si = sn[t * 32 + d]; o = s[d] * c - s[d + 32] * si; }
    else { int dd = d - 32; float c = cs[t * 32 + dd], si = sn[t * 32 + dd]; o = s[d] * c + s[dd] * si; }
    kr[((size_t)kv * T + t) * DH + d] = f2bf(o);
  } else {
    int kv = u - H - KVH;
    float o = qkv[(size_t)t * NQ + (H + KVH) * DH + kv * DH + d];
    vt[((size_t)kv * DH + d) * T + t] = f2bf(o);
  }
}

// ---------------- flash attention, 1 wave per (16-query tile, head) -----------
__global__ __launch_bounds__(32) void k_attn(const u16* __restrict__ qr,
                                             const u16* __restrict__ kr,
                                             const u16* __restrict__ vt,
                                             u16* __restrict__ attn) {
  int qt = blockIdx.x, hh = blockIdx.y, lane = threadIdx.x;
  int kv = hh >> 2;                       // GQA group = H/KV = 4
  int q0 = qt * 16;
  __shared__ __align__(16) u16 Pl[16 * 32];
  int row = lane & 15, k8 = (lane >> 4) << 3, kb16 = (lane >> 4) << 4, rb = (lane >> 4) << 3;
  Frag qa[2];
  { const u16* qrow = qr + ((size_t)hh * T + (q0 + row)) * DH;
#pragma unroll
    for (int c = 0; c < 2; c++) {
      qa[c].q[0] = *(const u32x4*)(qrow + c * 32 + kb16);
      qa[c].q[1] = *(const u32x4*)(qrow + c * 32 + kb16 + 8);
    } }
  v8f acc[4] = { v8zero(), v8zero(), v8zero(), v8zero() };
  float m[8], l[8];
#pragma unroll
  for (int r = 0; r < 8; r++) { m[r] = -1e30f; l[r] = 0.f; }
  const float scale = 0.125f;             // DH^-0.5
  for (int kb = 0; kb < q0 + 16; kb += 32) {
    v8f s0 = v8zero(), s1 = v8zero();
#pragma unroll
    for (int c = 0; c < 2; c++) {
      { const u16* krow = kr + ((size_t)kv * T + (kb + (lane & 15))) * DH + c * 32 + kb16;
        Frag fb; fb.q[0] = *(const u32x4*)krow; fb.q[1] = *(const u32x4*)(krow + 8);
        s0 = wmma_bf(qa[c], fb, s0); }
      { const u16* krow = kr + ((size_t)kv * T + (kb + 16 + (lane & 15))) * DH + c * 32 + kb16;
        Frag fb; fb.q[0] = *(const u32x4*)krow; fb.q[1] = *(const u32x4*)(krow + 8);
        s1 = wmma_bf(qa[c], fb, s1); }
    }
    int c0 = kb + (lane & 15);
#pragma unroll
    for (int r = 0; r < 8; r++) {
      int qi = q0 + rb + r;
      float a0 = (c0      <= qi) ? s0[r] * scale : -1e30f;
      float a1 = (c0 + 16 <= qi) ? s1[r] * scale : -1e30f;
      float mx = redmax16(fmaxf(a0, a1));
      float mn = fmaxf(m[r], mx);
      float alpha = __expf(m[r] - mn);
      float p0 = __expf(a0 - mn), p1 = __expf(a1 - mn);
      float rs = redsum16(p0 + p1);
      l[r] = l[r] * alpha + rs;
      m[r] = mn;
#pragma unroll
      for (int j = 0; j < 4; j++) acc[j][r] = acc[j][r] * alpha;
      Pl[(rb + r) * 32 + (lane & 15)] = f2bf(p0);
      Pl[(rb + r) * 32 + 16 + (lane & 15)] = f2bf(p1);
    }
    __syncthreads();
    Frag pa; pa.q[0] = *(const u32x4*)&Pl[row * 32 + kb16];
    pa.q[1] = *(const u32x4*)&Pl[row * 32 + kb16 + 8];
#pragma unroll
    for (int j = 0; j < 4; j++) {
      const u16* vrow = vt + ((size_t)kv * DH + (j * 16 + (lane & 15))) * T + kb + kb16;
      Frag fb; fb.q[0] = *(const u32x4*)vrow; fb.q[1] = *(const u32x4*)(vrow + 8);
      acc[j] = wmma_bf(pa, fb, acc[j]);
    }
    __syncthreads();
  }
#pragma unroll
  for (int j = 0; j < 4; j++)
#pragma unroll
    for (int r = 0; r < 8; r++) {
      float val = acc[j][r] / l[r];
      attn[(size_t)(q0 + rb + r) * D + hh * DH + j * 16 + (lane & 15)] = f2bf(val);
    }
}

// ---------------- router: logits + softmax + top-2 ---------------------------
__global__ __launch_bounds__(256) void k_router(const u16* __restrict__ x2,
                                                const float* __restrict__ gw,
                                                float* __restrict__ topw,
                                                int* __restrict__ topi) {
  int t = blockIdx.x, tid = threadIdx.x, lane = tid & 31, w = tid >> 5;
  float p[NE];
#pragma unroll
  for (int e = 0; e < NE; e++) p[e] = 0.f;
  for (int d = tid; d < D; d += 256) {
    float xv = bf2f(x2[(size_t)t * D + d]);
#pragma unroll
    for (int e = 0; e < NE; e++) p[e] += xv * gw[(size_t)e * D + d];
  }
#pragma unroll
  for (int e = 0; e < NE; e++) {
    p[e] = redsum16(p[e]);
    p[e] += __shfl_xor(p[e], 16, 32);
  }
  __shared__ float red[8][NE];
  if (lane == 0)
#pragma unroll
    for (int e = 0; e < NE; e++) red[w][e] = p[e];
  __syncthreads();
  if (tid == 0) {
    float lg[NE];
#pragma unroll
    for (int e = 0; e < NE; e++) { float s = 0; for (int ww = 0; ww < 8; ww++) s += red[ww][e]; lg[e] = s; }
    float mx = lg[0];
    for (int e = 1; e < NE; e++) mx = fmaxf(mx, lg[e]);
    float pr[NE]; float sum = 0;
    for (int e = 0; e < NE; e++) { pr[e] = __expf(lg[e] - mx); sum += pr[e]; }
    for (int e = 0; e < NE; e++) pr[e] /= sum;
    int e0 = 0; for (int e = 1; e < NE; e++) if (pr[e] > pr[e0]) e0 = e;
    int e1 = (e0 == 0) ? 1 : 0;
    for (int e = 0; e < NE; e++) if (e != e0 && pr[e] > pr[e1]) e1 = e;
    float s2 = pr[e0] + pr[e1];
    topi[t * 2] = e0; topi[t * 2 + 1] = e1;
    topw[t * 2] = pr[e0] / s2; topw[t * 2 + 1] = pr[e1] / s2;
  }
}

// -------- deterministic token->expert lists via single-block prefix scans ----
__global__ __launch_bounds__(512) void k_build(const int* __restrict__ topi,
                                               int* __restrict__ list,
                                               int* __restrict__ rowof,
                                               int* __restrict__ cntp,
                                               int* __restrict__ basep) {
  __shared__ int scan[T];
  __shared__ int sbase;
  int t = threadIdx.x;
  int e0 = topi[t * 2], e1 = topi[t * 2 + 1];
  if (t == 0) sbase = 0;
  __syncthreads();
  for (int e = 0; e < NE; e++) {
    int slot = (e0 == e) ? 0 : ((e1 == e) ? 1 : -1);
    int f = (slot >= 0) ? 1 : 0;
    scan[t] = f; __syncthreads();
    for (int off = 1; off < T; off <<= 1) {
      int v = (t >= off) ? scan[t - off] : 0;
      __syncthreads();
      scan[t] += v;
      __syncthreads();
    }
    int pos = scan[t] - f;
    int c = scan[T - 1];
    int cp = (c + 15) & ~15;
    int b = sbase;
    if (slot >= 0) { list[e * T + pos] = t; rowof[t * 2 + slot] = b + pos; }
    if (t >= c && t < cp) list[e * T + t] = 0;   // pad rows with token 0 (weight 0)
    __syncthreads();
    if (t == 0) { cntp[e] = cp; basep[e] = b; sbase = b + cp; }
    __syncthreads();
  }
}

// ---------------- MoE phase 1: act = silu(x2 @ w1e^T) * (x2 @ w3e^T) ----------
__global__ __launch_bounds__(256) void k_moe1(const u16* __restrict__ x2,
                                              const float* __restrict__ w1,
                                              const float* __restrict__ w3,
                                              const int* __restrict__ list,
                                              const int* __restrict__ cntp,
                                              const int* __restrict__ basep,
                                              u16* __restrict__ act) {
  int e = blockIdx.x >> 5, mt = blockIdx.x & 31;
  if (mt * 16 >= cntp[e]) return;
  int n0 = blockIdx.y * 256;
  int tid = threadIdx.x, lane = tid & 31, w = tid >> 5;
  const float* B1 = w1 + (size_t)e * FF * D;
  const float* B3 = w3 + (size_t)e * FF * D;
  __shared__ __align__(16) u16 AT[16 * 32];
  __shared__ __align__(16) u16 BT1[256 * 32];
  __shared__ __align__(16) u16 BT3[256 * 32];
  __shared__ int toks[16];
  if (tid < 16) toks[tid] = list[e * T + mt * 16 + tid];
  __syncthreads();
  v8f a1[2] = { v8zero(), v8zero() }, a3[2] = { v8zero(), v8zero() };
  for (int k0 = 0; k0 < D; k0 += 32) {
    { int ee = tid * 2, r = ee >> 5, kk = ee & 31;
      async_cp_b32(&x2[(size_t)toks[r] * D + k0 + kk], &AT[r * 32 + kk]); }
    { const float* s1 = B1 + (size_t)(n0 + tid) * D + k0;
      const float* s3 = B3 + (size_t)(n0 + tid) * D + k0;
      if (k0 + 32 < D) { __builtin_prefetch(s1 + 32, 0, 1); __builtin_prefetch(s3 + 32, 0, 1); }
#pragma unroll
      for (int i = 0; i < 8; i++) {
        float4 f = *(const float4*)(s1 + i * 4);
        u32x2 p; p.x = (u32)f2bf(f.x) | ((u32)f2bf(f.y) << 16);
        p.y = (u32)f2bf(f.z) | ((u32)f2bf(f.w) << 16);
        *(u32x2*)&BT1[tid * 32 + i * 4] = p;
        float4 g = *(const float4*)(s3 + i * 4);
        u32x2 q2; q2.x = (u32)f2bf(g.x) | ((u32)f2bf(g.y) << 16);
        q2.y = (u32)f2bf(g.z) | ((u32)f2bf(g.w) << 16);
        *(u32x2*)&BT3[tid * 32 + i * 4] = q2;
      } }
    async_wait();
    __syncthreads();
    int row = lane & 15, k8 = (lane >> 4) << 3, kb16 = (lane >> 4) << 4;
    Frag fa; fa.q[0] = *(const u32x4*)&AT[row * 32 + kb16];
    fa.q[1] = *(const u32x4*)&AT[row * 32 + kb16 + 8];
#pragma unroll
    for (int c = 0; c < 2; c++) {
      int n = w * 32 + c * 16 + (lane & 15);
      Frag f1; f1.q[0] = *(const u32x4*)&BT1[n * 32 + kb16];
      f1.q[1] = *(const u32x4*)&BT1[n * 32 + kb16 + 8];
      a1[c] = wmma_bf(fa, f1, a1[c]);
      Frag f3; f3.q[0] = *(const u32x4*)&BT3[n * 32 + kb16];
      f3.q[1] = *(const u32x4*)&BT3[n * 32 + kb16 + 8];
      a3[c] = wmma_bf(fa, f3, a3[c]);
    }
    __syncthreads();
  }
  int rb = (lane >> 4) << 3;
  int g0 = basep[e] + mt * 16;
#pragma unroll
  for (int c = 0; c < 2; c++)
#pragma unroll
    for (int r = 0; r < 8; r++) {
      int grow = g0 + rb + r, gn = n0 + w * 32 + c * 16 + (lane & 15);
      float h1 = a1[c][r], h3 = a3[c][r];
      float sv = h1 / (1.f + __expf(-h1));
      act[(size_t)grow * FF + gn] = f2bf(sv * h3);
    }
}

// ---------------- MoE phase 2: eout = act @ w2e^T -----------------------------
__global__ __launch_bounds__(256) void k_moe2(const u16* __restrict__ act,
                                              const float* __restrict__ w2,
                                              const int* __restrict__ cntp,
                                              const int* __restrict__ basep,
                                              float* __restrict__ eout) {
  int e = blockIdx.x >> 5, mt = blockIdx.x & 31;
  if (mt * 16 >= cntp[e]) return;
  int n0 = blockIdx.y * 256;
  int tid = threadIdx.x, lane = tid & 31, w = tid >> 5;
  const float* Bw = w2 + (size_t)e * D * FF;
  __shared__ __align__(16) u16 AT[16 * 32];
  __shared__ __align__(16) u16 BT[256 * 32];
  int g0 = basep[e] + mt * 16;
  v8f acc[2] = { v8zero(), v8zero() };
  for (int k0 = 0; k0 < FF; k0 += 32) {
    { int ee = tid * 2, r = ee >> 5, kk = ee & 31;
      async_cp_b32(&act[(size_t)(g0 + r) * FF + k0 + kk], &AT[r * 32 + kk]); }
    { const float* src = Bw + (size_t)(n0 + tid) * FF + k0;
      if (k0 + 32 < FF) __builtin_prefetch(src + 32, 0, 1);
#pragma unroll
      for (int i = 0; i < 8; i++) {
        float4 f = *(const float4*)(src + i * 4);
        u32x2 p; p.x = (u32)f2bf(f.x) | ((u32)f2bf(f.y) << 16);
        p.y = (u32)f2bf(f.z) | ((u32)f2bf(f.w) << 16);
        *(u32x2*)&BT[tid * 32 + i * 4] = p;
      } }
    async_wait();
    __syncthreads();
    int row = lane & 15, k8 = (lane >> 4) << 3, kb16 = (lane >> 4) << 4;
    Frag fa; fa.q[0] = *(const u32x4*)&AT[row * 32 + kb16];
    fa.q[1] = *(const u32x4*)&AT[row * 32 + kb16 + 8];
#pragma unroll
    for (int c = 0; c < 2; c++) {
      int n = w * 32 + c * 16 + (lane & 15);
      Frag fb; fb.q[0] = *(const u32x4*)&BT[n * 32 + kb16];
      fb.q[1] = *(const u32x4*)&BT[n * 32 + kb16 + 8];
      acc[c] = wmma_bf(fa, fb, acc[c]);
    }
    __syncthreads();
  }
  int rb = (lane >> 4) << 3;
#pragma unroll
  for (int c = 0; c < 2; c++)
#pragma unroll
    for (int r = 0; r < 8; r++) {
      int gm = g0 + rb + r, gn = n0 + w * 32 + c * 16 + (lane & 15);
      eout[(size_t)gm * D + gn] = acc[c][r];
    }
}

// ---------------- weighted top-2 combine -------------------------------------
__global__ __launch_bounds__(256) void k_combine(const float* __restrict__ eout,
                                                 const int* __restrict__ rowof,
                                                 const float* __restrict__ topw,
                                                 float* __restrict__ out) {
  int t = blockIdx.x; int d = blockIdx.y * 256 + threadIdx.x;
  int r0 = rowof[t * 2], r1 = rowof[t * 2 + 1];
  float w0 = topw[t * 2], w1 = topw[t * 2 + 1];
  out[(size_t)t * D + d] = w0 * eout[(size_t)r0 * D + d] + w1 * eout[(size_t)r1 * D + d];
}

extern "C" void kernel_launch(void* const* d_in, const int* in_sizes, int n_in,
                              void* d_out, int out_size, void* d_ws, size_t ws_size,
                              hipStream_t stream) {
  const float* hs   = (const float*)d_in[0];
  const float* rsd  = (const float*)d_in[1];
  const float* cs   = (const float*)d_in[2];
  const float* sn   = (const float*)d_in[3];
  const float* ln1  = (const float*)d_in[4];
  const float* ln2  = (const float*)d_in[5];
  const float* wqkv = (const float*)d_in[6];
  const float* wo   = (const float*)d_in[7];
  const float* gw   = (const float*)d_in[8];
  const float* w1   = (const float*)d_in[9];
  const float* w3   = (const float*)d_in[10];
  const float* w2   = (const float*)d_in[11];

  float* moe_out = (float*)d_out;
  float* res2    = (float*)d_out + (size_t)T * D;

  char* ws = (char*)d_ws;
  float* res1 = (float*)(ws + OFF_RES1);
  u16*   x1   = (u16*)(ws + OFF_X1);
  float* qkv  = (float*)(ws + OFF_QKV);
  u16*   qr   = (u16*)(ws + OFF_QR);
  u16*   kr   = (u16*)(ws + OFF_KR);
  u16*   vt   = (u16*)(ws + OFF_VT);
  u16*   attn = (u16*)(ws + OFF_ATT);
  u16*   x2   = (u16*)(ws + OFF_X2);
  float* topw = (float*)(ws + OFF_TW);
  int*   topi = (int*)(ws + OFF_TI);
  int*   rowof= (int*)(ws + OFF_RO);
  int*   list = (int*)(ws + OFF_LS);
  int*   cntp = (int*)(ws + OFF_CP);
  int*   basep= (int*)(ws + OFF_BS);
  u16*   act  = (u16*)(ws + OFF_ACT);
  float* eout = (float*)(ws + OFF_EO);

  k_addnorm<<<T, 256, 0, stream>>>(hs, rsd, ln1, res1, x1);
  k_gemm<0><<<dim3(T / 16, NQ / 256), 256, 0, stream>>>(x1, wqkv, qkv, nullptr, D, NQ);
  k_rope<<<dim3(T, H + 2 * KVH), 64, 0, stream>>>(qkv, cs, sn, qr, kr, vt);
  k_attn<<<dim3(T / 16, H), 32, 0, stream>>>(qr, kr, vt, attn);
  k_gemm<1><<<dim3(T / 16, D / 256), 256, 0, stream>>>(attn, wo, res2, res1, D, D);
  k_addnorm<<<T, 256, 0, stream>>>(res2, nullptr, ln2, nullptr, x2);
  k_router<<<T, 256, 0, stream>>>(x2, gw, topw, topi);
  k_build<<<1, T, 0, stream>>>(topi, list, rowof, cntp, basep);
  k_moe1<<<dim3(NE * 32, FF / 256), 256, 0, stream>>>(x2, w1, w3, list, cntp, basep, act);
  k_moe2<<<dim3(NE * 32, D / 256), 256, 0, stream>>>(act, w2, cntp, basep, eout);
  k_combine<<<dim3(T, D / 256), 256, 0, stream>>>(eout, rowof, topw, moe_out);
}
